// PottsSequenceModel_65335042506803
// MI455X (gfx1250) — matrix operand order repaired and checked by reading, not compiled, not run on hardware
//
#include <hip/hip_runtime.h>
#include <hip/hip_bf16.h>
#include <cstdint>
#include <cstddef>

#define NRES 8192
#define KNN  32
#define AA   20
#define PPW  8   // pairs per wave in k2 pipeline

typedef __attribute__((ext_vector_type(16))) _Float16 v16h;
typedef __attribute__((ext_vector_type(8)))  float    v8f;
typedef __attribute__((ext_vector_type(4)))  float    f4;

// ---------------- wave helpers (wave32) ----------------
__device__ __forceinline__ float wave_max20(float v, int lane) {
  float x = (lane < AA) ? v : -3.4e38f;
  #pragma unroll
  for (int off = 16; off > 0; off >>= 1)
    x = fmaxf(x, __shfl_xor(x, off, 32));
  return x;
}
__device__ __forceinline__ float wave_sum(float v) {
  #pragma unroll
  for (int off = 16; off > 0; off >>= 1)
    v += __shfl_xor(v, off, 32);
  return v;
}

// ---------------- k0: one-hot aa -> index ----------------
__global__ void k0_prep(const float* __restrict__ aa, int* __restrict__ aa_idx) {
  int i = blockIdx.x * blockDim.x + threadIdx.x;
  if (i >= NRES) return;
  int idx = 0;
  #pragma unroll
  for (int a = 0; a < AA; ++a)
    if (aa[i * AA + a] > 0.5f) idx = a;
  aa_idx[i] = idx;
}

// ---------------- k1: r_i = h_masked + sum_j J[:, b_j] ----------------
__global__ void __launch_bounds__(256) k1_r(const float* __restrict__ h,
                                            const float* __restrict__ J,
                                            const float* __restrict__ mask,
                                            const int*   __restrict__ nbr,
                                            const int*   __restrict__ aa_idx,
                                            float*       __restrict__ r) {
  __shared__ __align__(16) float Jlds[8][400];
  int wave = threadIdx.x >> 5;
  int lane = threadIdx.x & 31;
  int i = blockIdx.x * 8 + wave;
  if (i >= NRES) return;
  float mi = mask[i];
  float acc = 0.f;
  const f4* Jv = (const f4*)J;
  f4* dst = (f4*)&Jlds[wave][0];
  for (int j = 0; j < KNN; ++j) {
    int p = i * KNN + j;
    int n = nbr[p];
    bool act = (n >= 0) && (mi > 0.f);
    if (act) act = (mask[n] > 0.f);
    if (!act) continue;                       // wave-uniform: skip masked pair's block
    long long b4 = (long long)p * 100;        // 100 float4 per 20x20 block
    #pragma unroll
    for (int k = 0; k < 3; ++k)
      dst[k * 32 + lane] = __builtin_nontemporal_load(&Jv[b4 + k * 32 + lane]);
    if (lane < 4)
      dst[96 + lane] = __builtin_nontemporal_load(&Jv[b4 + 96 + lane]);
    asm volatile("s_wait_dscnt 0" ::: "memory");
    int bj = aa_idx[n];
    if (lane < AA) acc += Jlds[wave][lane * AA + bj];
  }
  if (lane < AA)
    r[i * AA + lane] = (mi > 0.f ? h[i * AA + lane] : 0.f) + acc;
}

// ---------------- async prefetch of one 1600B J block into LDS (NT: stream, don't thrash L2) ----
__device__ __forceinline__ void issue_prefetch(const float* J, long long p,
                                               float* dstLds, int lane) {
  unsigned long long ga = (unsigned long long)(const void*)((const f4*)J + p * 100 + lane);
  unsigned lo = (unsigned)(size_t)(void*)((f4*)dstLds + lane);
  #pragma unroll
  for (int k = 0; k < 3; ++k) {
    asm volatile("global_load_async_to_lds_b128 %0, %1, off th:TH_LOAD_NT"
                 :: "v"(lo + (unsigned)(k * 512)),
                    "v"(ga + (unsigned long long)(k * 512))
                 : "memory");
  }
  if (lane < 4) {
    asm volatile("global_load_async_to_lds_b128 %0, %1, off th:TH_LOAD_NT"
                 :: "v"(lo + 1536u), "v"(ga + 1536ull) : "memory");
  }
  // 4 instructions issued -> ASYNCcnt += 4
}

// ---------------- per-pair score: logsumexp via 2x WMMA f16 matvec ----------------
// scr layout: [0..19]=eu, [20..31]=0, [32..51]=ev, [52..63]=0  (pads written once per wave)
__device__ __forceinline__ float pair_score(const float* __restrict__ Jb, // LDS, 400 f32
                                            float* __restrict__ scr,      // LDS, 64 f32
                                            const float* __restrict__ r,
                                            int i, int n, int ai, int bj, int lane) {
  const int half = lane >> 4;
  const int lr   = lane & 15;
  const int lc   = (lane < AA) ? lane : (AA - 1);

  float u = r[i * AA + lc]            - Jb[lc * AA + bj];    // u[a], lane a
  float v = r[(long long)n * AA + lc] - Jb[ai * AA + lc];    // v[b], lane b
  float mu = wave_max20(u, lane);
  float mv = wave_max20(v, lane);
  float eu = __expf(u - mu);
  float ev = __expf(v - mv);
  if (lane < AA) { scr[lane] = eu; scr[32 + lane] = ev; }
  asm volatile("s_wait_dscnt 0" ::: "memory");

  // B fragment: column 0 = ev (K=b, padded to 32).  element e <-> K = e + 16*half, N = lr
  v16h bf;
  #pragma unroll
  for (int e = 0; e < 16; ++e) {
    int kk = e + 16 * half;                 // 0..31, scr pad makes this safe
    float evk = scr[32 + kk];               // unconditional (vectorizes to ds_load_b128)
    bf[e] = (_Float16)((lr == 0) ? evk : 0.f);
  }
  // A fragments: E = exp(J).  element e <-> K = e + 16*half; rows lr / 16+lr
  const int   row2  = 16 + lr;
  const bool  r2ok  = (row2 < AA);
  const float* Jr1  = Jb + lr * AA;
  const float* Jr2  = Jb + (r2ok ? row2 : 0) * AA;
  v16h a1, a2;
  #pragma unroll
  for (int e = 0; e < 16; ++e) {
    int  kk  = e + 16 * half;
    bool kok = (kk < AA);
    int  kc  = kok ? kk : 0;
    float e1 = __expf(Jr1[kc]);
    float e2 = __expf(Jr2[kc]);
    a1[e] = (_Float16)(kok ? e1 : 0.f);
    a2[e] = (_Float16)((kok && r2ok) ? e2 : 0.f);
  }

  v8f d1 = {}, d2 = {};
  d1 = __builtin_amdgcn_wmma_f32_16x16x32_f16(false, a1, false, bf, (short)0, d1, false, false);
  d2 = __builtin_amdgcn_wmma_f32_16x16x32_f16(false, a2, false, bf, (short)0, d2, false, false);

  // Z = eu . t ; D layout: VGPR rr <-> M = rr + 8*half; tile2 rows map to eu[16..31] (pad=0)
  float s = 0.f;
  #pragma unroll
  for (int rr = 0; rr < 8; ++rr)
    s += d1[rr] * scr[rr + 8 * half];       // 0..15
  #pragma unroll
  for (int rr = 0; rr < 8; ++rr)
    s += d2[rr] * scr[16 + rr + 8 * half];  // 16..31, zeros beyond 19
  float Z = wave_sum(s);

  float uai = __shfl(u, ai, 32);
  float vbj = __shfl(v, bj, 32);
  return (uai - mu) + (vbj - mv) + Jb[ai * AA + bj] - __logf(Z);
}

// ---------------- k2: per-pair log_p_ij, async double-buffered ----------------
__global__ void __launch_bounds__(256) k2_score(const float* __restrict__ J,
                                                const float* __restrict__ mask,
                                                const int*   __restrict__ nbr,
                                                const int*   __restrict__ aa_idx,
                                                const float* __restrict__ r,
                                                float*       __restrict__ lp) {
  __shared__ __align__(16) float Jbuf[8][2][400];
  __shared__ __align__(16) float scr[8][64];
  int wave = threadIdx.x >> 5;
  int lane = threadIdx.x & 31;
  long long base = ((long long)blockIdx.x * 8 + wave) * PPW;

  float* myJ[2] = { &Jbuf[wave][0][0], &Jbuf[wave][1][0] };
  float* myscr  = &scr[wave][0];
  if (lane >= AA) { myscr[lane] = 0.f; myscr[32 + lane] = 0.f; }  // zero pads, once

  int  n_cur;  bool a_cur;
  {
    int ii = (int)(base >> 5);
    n_cur = nbr[base];
    a_cur = (n_cur >= 0) && (mask[ii] > 0.f) && (mask[n_cur] > 0.f);
  }
  if (a_cur) issue_prefetch(J, base, myJ[0], lane);

  for (int t = 0; t < PPW; ++t) {
    long long p = base + t;
    int n_nxt = 0; bool a_nxt = false;
    if (t + 1 < PPW) {
      long long pn = p + 1;
      int ii = (int)(pn >> 5);
      n_nxt = nbr[pn];
      a_nxt = (n_nxt >= 0) && (mask[ii] > 0.f) && (mask[n_nxt] > 0.f);
      if (a_nxt) issue_prefetch(J, pn, myJ[(t + 1) & 1], lane);
    }
    float out = 0.f;
    if (a_cur) {                                   // wave-uniform branch: EXEC full for WMMA
      if (a_nxt) asm volatile("s_wait_asynccnt 4" ::: "memory");
      else       asm volatile("s_wait_asynccnt 0" ::: "memory");
      int ii = (int)(p >> 5);
      out = pair_score(myJ[t & 1], myscr, r, ii, n_cur, aa_idx[ii], aa_idx[n_cur], lane);
    }
    if (lane == 0) lp[p] = out;
    n_cur = n_nxt; a_cur = a_nxt;
  }
}

// ---------------- k3: log_p_i reduction ----------------
__global__ void k3_final(const float* __restrict__ mask,
                         const int*   __restrict__ nbr,
                         const float* __restrict__ lp,
                         float*       __restrict__ out) {
  int i = blockIdx.x * blockDim.x + threadIdx.x;
  if (i >= NRES) return;
  float mi = mask[i];
  float s = 0.f, c = 0.f;
  for (int j = 0; j < KNN; ++j) {
    int p = i * KNN + j;
    int n = nbr[p];
    if (n >= 0 && mi > 0.f && mask[n] > 0.f) c += 1.f;
    s += lp[p];                                   // already 0 where pair_mask==0
  }
  out[i] = mi * s / fmaxf(2.f * c, 2.f);
}

extern "C" void kernel_launch(void* const* d_in, const int* in_sizes, int n_in,
                              void* d_out, int out_size, void* d_ws, size_t ws_size,
                              hipStream_t stream) {
  const float* h    = (const float*)d_in[0];
  const float* J    = (const float*)d_in[1];
  const float* aa   = (const float*)d_in[2];
  const float* mask = (const float*)d_in[3];
  const int*   nbr  = (const int*)d_in[4];

  float* log_p_i  = (float*)d_out;              // [N]
  float* log_p_ij = (float*)d_out + NRES;       // [N, K]

  float* r      = (float*)d_ws;                                   // N*AA floats
  int*   aa_idx = (int*)((char*)d_ws + (size_t)NRES * AA * sizeof(float));

  k0_prep<<<NRES / 256, 256, 0, stream>>>(aa, aa_idx);
  k1_r   <<<NRES / 8,   256, 0, stream>>>(h, J, mask, nbr, aa_idx, r);
  k2_score<<<(NRES * KNN) / (8 * PPW), 256, 0, stream>>>(J, mask, nbr, aa_idx, r, log_p_ij);
  k3_final<<<NRES / 256, 256, 0, stream>>>(mask, nbr, log_p_ij, log_p_i);
}
